// pGNN_58858231824471
// MI455X (gfx1250) — matrix operand order repaired and checked by reading, not compiled, not run on hardware
//
#include <hip/hip_runtime.h>
#include <hip/hip_fp16.h>

typedef __attribute__((ext_vector_type(16))) _Float16 v16h;
typedef __attribute__((ext_vector_type(8)))  _Float16 v8h;
typedef __attribute__((ext_vector_type(4)))  _Float16 v4h;
typedef __attribute__((ext_vector_type(8)))  float    v8f;

#define MU_C   0.1f
#define P_C    2.5f
#define HID    64
#define IN_DIM 256
#define OUT_DIM 40

// ---------------- degree / inv-sqrt ----------------
__global__ void deg_kernel(const long long* __restrict__ col, float* __restrict__ deg, int E) {
  int e = blockIdx.x * blockDim.x + threadIdx.x;
  if (e < E) atomicAdd(&deg[(int)col[e]], 1.0f);
}

__global__ void dis_kernel(const float* __restrict__ deg, float* __restrict__ dis, int N) {
  int n = blockIdx.x * blockDim.x + threadIdx.x;
  if (n < N) {
    float d = deg[n];
    dis[n] = (d > 0.0f) ? rsqrtf(d) : 0.0f;
  }
}

// ---------------- GEMM1: h = relu(x @ W1 + b1), also seed out = h --------
// 4 waves/block; wave w computes rows [blockIdx*16,+16) x cols [16w,16w+16).
// Shared A tile (16x256 of x) staged once per block in LDS as f16; each wave
// then assembles its WMMA A fragment from two aligned ds_load_b128s.
__global__ __launch_bounds__(128) void gemm1_kernel(
    const float* __restrict__ x, const float* __restrict__ W1,
    const float* __restrict__ b1, float* __restrict__ h,
    float* __restrict__ outA, int N) {
  __shared__ __attribute__((aligned(16))) _Float16 sX[16 * IN_DIM];  // 8 KB

  const int lane    = threadIdx.x & 31;
  const int wave    = threadIdx.x >> 5;
  const int rowBase = blockIdx.x * 16;
  const int colBase = wave * 16;

  // ---- cooperative staging: 16 rows x 64 float4 = 1024 chunks, 8/thread ----
#pragma unroll
  for (int it = 0; it < 8; ++it) {
    int idx = threadIdx.x + it * 128;
    int r   = idx >> 6;            // 0..15
    int c4  = idx & 63;            // 0..63 float4 within row
    int gr  = rowBase + r; if (gr >= N) gr = N - 1;
    float4 f = ((const float4*)(x + (size_t)gr * IN_DIM))[c4];
    v4h hv = { (_Float16)f.x, (_Float16)f.y, (_Float16)f.z, (_Float16)f.w };
    *(v4h*)&sX[r * IN_DIM + c4 * 4] = hv;
  }
  __syncthreads();

  const int m   = lane & 15;
  const int kb  = (lane >> 4) * 8;    // A-matrix K interleave per half-wave
  const int kgB = (lane >> 4) * 16;   // unused; B below uses kb, same K order as A
  const int bcol = colBase + m;

  v8f c = {};
  for (int k0 = 0; k0 < IN_DIM; k0 += 32) {
    // A fragment: halves 0-7 -> K = k0+kb+0..7 ; halves 8-15 -> K = k0+16+kb+0..7
    v8h alo = *(const v8h*)&sX[m * IN_DIM + k0 + kb];
    v8h ahi = *(const v8h*)&sX[m * IN_DIM + k0 + 16 + kb];
    v16h a = __builtin_shufflevector(alo, ahi,
                                     0, 1, 2, 3, 4, 5, 6, 7,
                                     8, 9, 10, 11, 12, 13, 14, 15);
    v16h b;
#pragma unroll
    for (int i = 0; i < 16; ++i) {
      int Kb = k0 + kb + i + (i >= 8 ? 8 : 0);        // ISA 16-bit B 32x16 layout
      b[i] = (_Float16)W1[Kb * HID + bcol];
    }
    c = __builtin_amdgcn_wmma_f32_16x16x32_f16(false, a, false, b,
                                               (short)0, c, false, false);
  }
  // C/D layout: lane n = lane&15 ; VGPR r -> row r + 8*(lane>>4)
  const int colOut = colBase + (lane & 15);
  const int rOff   = (lane >> 4) * 8;
  const float bias = b1[colOut];
  if (rowBase + 16 <= N) {            // full-tile fast path: no per-row guards
#pragma unroll
    for (int r = 0; r < 8; ++r) {
      int rr = rowBase + rOff + r;
      float v = c[r] + bias;
      v = v > 0.0f ? v : 0.0f;
      size_t idx = (size_t)rr * HID + colOut;
      h[idx]    = v;
      outA[idx] = v;
    }
  } else {
#pragma unroll
    for (int r = 0; r < 8; ++r) {
      int rr = rowBase + rOff + r;
      if (rr < N) {
        float v = c[r] + bias;
        v = v > 0.0f ? v : 0.0f;
        size_t idx = (size_t)rr * HID + colOut;
        h[idx]    = v;
        outA[idx] = v;
      }
    }
  }
}

// ---------------- edge pass 1: M[e] = ||dis_r*out_r - dis_c*out_c||^(P-2) ----
// 8 lanes per edge, each lane covers 8 of the 64 features (2 x float4)
__global__ void calcM_kernel(const long long* __restrict__ row,
                             const long long* __restrict__ col,
                             const float* __restrict__ dis,
                             const float* __restrict__ outC,
                             float* __restrict__ M, float* __restrict__ Msum, int E) {
  long long tid = (long long)blockIdx.x * blockDim.x + threadIdx.x;
  int e   = (int)(tid >> 3);
  int sub = (int)(tid & 7);
  if (e >= E) return;
  int r = (int)row[e], c = (int)col[e];
  float dr = dis[r], dc = dis[c];
  const float4* pr = (const float4*)(outC + (size_t)r * HID) + sub * 2;
  const float4* pc = (const float4*)(outC + (size_t)c * HID) + sub * 2;
  float acc = 0.0f;
#pragma unroll
  for (int t = 0; t < 2; ++t) {
    float4 a = pr[t], b = pc[t];
    float d0 = dr * a.x - dc * b.x;
    float d1 = dr * a.y - dc * b.y;
    float d2 = dr * a.z - dc * b.z;
    float d3 = dr * a.w - dc * b.w;
    acc += d0 * d0 + d1 * d1 + d2 * d2 + d3 * d3;
  }
  acc += __shfl_xor(acc, 1, 8);
  acc += __shfl_xor(acc, 2, 8);
  acc += __shfl_xor(acc, 4, 8);
  if (sub == 0) {
    float norm = sqrtf(acc);
    float Mv   = sqrtf(norm);           // norm^(P-2) with P=2.5
    if (!__builtin_isfinite(Mv)) Mv = 0.0f;
    M[e] = Mv;
    atomicAdd(&Msum[c], Mv);
  }
}

// ---------------- alpha/beta + seed out_next = beta * h ----------------
__global__ void alpha_init_kernel(const float* __restrict__ dis,
                                  const float* __restrict__ Msum,
                                  const float* __restrict__ h,
                                  float* __restrict__ alpha,
                                  float* __restrict__ outN, int N) {
  long long tid = (long long)blockIdx.x * blockDim.x + threadIdx.x;
  int n = (int)(tid >> 4);
  int j = (int)(tid & 15);
  if (n >= N) return;
  float d = dis[n];
  float a = 1.0f / (d * d * Msum[n] + 2.0f * MU_C / P_C);
  float beta = (4.0f * MU_C / P_C) * a;
  if (j == 0) alpha[n] = a;
  float4 v = ((const float4*)(h + (size_t)n * HID))[j];
  ((float4*)(outN + (size_t)n * HID))[j] =
      make_float4(beta * v.x, beta * v.y, beta * v.z, beta * v.w);
}

// ---------------- propagate: out_next[c] += alpha[r]*dis[r]*M*dis[c]*out[r] ---
__global__ void propagate_kernel(const long long* __restrict__ row,
                                 const long long* __restrict__ col,
                                 const float* __restrict__ dis,
                                 const float* __restrict__ alpha,
                                 const float* __restrict__ M,
                                 const float* __restrict__ outC,
                                 float* __restrict__ outN, int E) {
  long long tid = (long long)blockIdx.x * blockDim.x + threadIdx.x;
  int e   = (int)(tid >> 3);
  int sub = (int)(tid & 7);
  if (e >= E) return;
  int r = (int)row[e], c = (int)col[e];
  float s = alpha[r] * dis[r] * M[e] * dis[c];
  const float4* src = (const float4*)(outC + (size_t)r * HID) + sub * 2;
  float*        dst = outN + (size_t)c * HID + sub * 8;
  float4 a = src[0], b = src[1];
  atomicAdd(dst + 0, s * a.x);
  atomicAdd(dst + 1, s * a.y);
  atomicAdd(dst + 2, s * a.z);
  atomicAdd(dst + 3, s * a.w);
  atomicAdd(dst + 4, s * b.x);
  atomicAdd(dst + 5, s * b.y);
  atomicAdd(dst + 6, s * b.z);
  atomicAdd(dst + 7, s * b.w);
}

// ---------------- GEMM2: logits = out @ W2 + b2 (64 x 40, cols padded to 48) --
__global__ __launch_bounds__(96) void gemm2_kernel(
    const float* __restrict__ out, const float* __restrict__ W2,
    const float* __restrict__ b2, float* __restrict__ logits, int N) {
  __shared__ __attribute__((aligned(16))) _Float16 sA[16 * HID];  // 2 KB

  const int lane    = threadIdx.x & 31;
  const int wave    = threadIdx.x >> 5;       // 3 waves -> 48 padded cols
  const int rowBase = blockIdx.x * 16;
  const int colBase = wave * 16;

  // ---- stage A tile: 16 rows x 16 float4 = 256 chunks over 96 threads ----
#pragma unroll
  for (int it = 0; it < 3; ++it) {
    int idx = threadIdx.x + it * 96;
    if (idx < 256) {
      int r  = idx >> 4;          // 0..15
      int c4 = idx & 15;          // 0..15 float4 within row
      int gr = rowBase + r; if (gr >= N) gr = N - 1;
      float4 f = ((const float4*)(out + (size_t)gr * HID))[c4];
      v4h hv = { (_Float16)f.x, (_Float16)f.y, (_Float16)f.z, (_Float16)f.w };
      *(v4h*)&sA[r * HID + c4 * 4] = hv;
    }
  }
  __syncthreads();

  const int m   = lane & 15;
  const int kb  = (lane >> 4) * 8;
  const int kgB = (lane >> 4) * 16;
  const int bcol = colBase + m;

  v8f c = {};
  for (int k0 = 0; k0 < HID; k0 += 32) {
    v8h alo = *(const v8h*)&sA[m * HID + k0 + kb];
    v8h ahi = *(const v8h*)&sA[m * HID + k0 + 16 + kb];
    v16h a = __builtin_shufflevector(alo, ahi,
                                     0, 1, 2, 3, 4, 5, 6, 7,
                                     8, 9, 10, 11, 12, 13, 14, 15);
    v16h b;
#pragma unroll
    for (int i = 0; i < 16; ++i) {
      int Kb = k0 + kb + i + (i >= 8 ? 8 : 0);
      b[i] = (_Float16)((bcol < OUT_DIM) ? W2[Kb * OUT_DIM + bcol] : 0.0f);
    }
    c = __builtin_amdgcn_wmma_f32_16x16x32_f16(false, a, false, b,
                                               (short)0, c, false, false);
  }
  const int colOut = colBase + (lane & 15);
  const int rOff   = (lane >> 4) * 8;
  if (colOut < OUT_DIM) {
    const float bias = b2[colOut];
    if (rowBase + 16 <= N) {
#pragma unroll
      for (int r = 0; r < 8; ++r) {
        int rr = rowBase + rOff + r;
        logits[(size_t)rr * OUT_DIM + colOut] = c[r] + bias;
      }
    } else {
#pragma unroll
      for (int r = 0; r < 8; ++r) {
        int rr = rowBase + rOff + r;
        if (rr < N) logits[(size_t)rr * OUT_DIM + colOut] = c[r] + bias;
      }
    }
  }
}

// ---------------- row-wise log_softmax over 40 logits, in place ----------
__global__ void logsoftmax_kernel(float* __restrict__ logits, int N) {
  int n = blockIdx.x * blockDim.x + threadIdx.x;
  if (n >= N) return;
  float* p = logits + (size_t)n * OUT_DIM;
  float mx = p[0];
#pragma unroll 8
  for (int j = 1; j < OUT_DIM; ++j) mx = fmaxf(mx, p[j]);
  float s = 0.0f;
#pragma unroll 8
  for (int j = 0; j < OUT_DIM; ++j) s += __expf(p[j] - mx);
  float lse = mx + __logf(s);
#pragma unroll 8
  for (int j = 0; j < OUT_DIM; ++j) p[j] -= lse;
}

extern "C" void kernel_launch(void* const* d_in, const int* in_sizes, int n_in,
                              void* d_out, int out_size, void* d_ws, size_t ws_size,
                              hipStream_t stream) {
  const float*     x   = (const float*)d_in[0];
  const long long* ei  = (const long long*)d_in[1];
  const float*     W1  = (const float*)d_in[2];
  const float*     b1  = (const float*)d_in[3];
  const float*     W2  = (const float*)d_in[4];
  const float*     b2  = (const float*)d_in[5];
  const int N = in_sizes[0] / IN_DIM;
  const int E = in_sizes[1] / 2;
  const long long* rowp = ei;       // edge_index[0]
  const long long* colp = ei + E;   // edge_index[1]

  // workspace layout (floats), 256B-aligned sections
  const size_t Np = ((size_t)N + 63) & ~(size_t)63;
  const size_t Ep = ((size_t)E + 63) & ~(size_t)63;
  float* deg   = (float*)d_ws;
  float* dis   = deg   + Np;
  float* Msum  = dis   + Np;
  float* alpha = Msum  + Np;
  float* M     = alpha + Np;
  float* h     = M     + Ep;
  float* outA  = h     + (size_t)N * HID;
  float* outB  = outA  + (size_t)N * HID;
  (void)ws_size; (void)n_in; (void)out_size;

  // 1) degrees and deg^{-1/2}
  hipMemsetAsync(deg, 0, (size_t)N * sizeof(float), stream);
  deg_kernel<<<(E + 255) / 256, 256, 0, stream>>>(colp, deg, E);
  dis_kernel<<<(N + 255) / 256, 256, 0, stream>>>(deg, dis, N);

  // 2) h = relu(x@W1+b1); out = h
  gemm1_kernel<<<(N + 15) / 16, 128, 0, stream>>>(x, W1, b1, h, outA, N);

  // 3) K=2 p-Laplacian iterations
  float* cur = outA;
  float* nxt = outB;
  const unsigned edgeBlocks = (unsigned)(((long long)E * 8 + 255) / 256);
  const unsigned nodeBlocks = (unsigned)(((long long)N * 16 + 255) / 256);
  for (int it = 0; it < 2; ++it) {
    hipMemsetAsync(Msum, 0, (size_t)N * sizeof(float), stream);
    calcM_kernel<<<edgeBlocks, 256, 0, stream>>>(rowp, colp, dis, cur, M, Msum, E);
    alpha_init_kernel<<<nodeBlocks, 256, 0, stream>>>(dis, Msum, h, alpha, nxt, N);
    propagate_kernel<<<edgeBlocks, 256, 0, stream>>>(rowp, colp, dis, alpha, M, cur, nxt, E);
    float* t = cur; cur = nxt; nxt = t;
  }

  // 4) logits + log_softmax -> d_out
  gemm2_kernel<<<(N + 15) / 16, 96, 0, stream>>>(cur, W2, b2, (float*)d_out, N);
  logsoftmax_kernel<<<(N + 255) / 256, 256, 0, stream>>>((float*)d_out, N);
}
